// GMLayer_48284022341700
// MI455X (gfx1250) — compile-verified, matrix operand order checked by reading, not run on hardware
//
#include <hip/hip_runtime.h>
#include <hip/hip_bf16.h>
#include <stdint.h>

#define NODES 8192
#define DIM   256
#define NHEADS 4
#define FOUT_  64
#define DSTATE 16
#define DTRANK 16

typedef __attribute__((ext_vector_type(16))) __bf16         v16bf;
typedef __attribute__((ext_vector_type(16))) unsigned short v16u;
typedef __attribute__((ext_vector_type(8)))  unsigned short u16x8;
typedef __attribute__((ext_vector_type(8)))  float          v8f;
typedef __attribute__((ext_vector_type(4)))  unsigned int   u32x4;
typedef __attribute__((ext_vector_type(8)))  int            i32x8;
typedef __attribute__((ext_vector_type(4)))  int            i32x4;

// ---------- helpers ----------
__device__ __forceinline__ unsigned short f2bf(float f) {
  unsigned u = __float_as_uint(f);
  u += 0x7FFFu + ((u >> 16) & 1u);           // round-to-nearest-even
  return (unsigned short)(u >> 16);
}
__device__ __forceinline__ float sigm(float x) { return 1.0f / (1.0f + __expf(-x)); }

__global__ void cvt_bf16(const float* __restrict__ in, unsigned short* __restrict__ out, int n) {
  int i = blockIdx.x * blockDim.x + threadIdx.x;
  if (i < n) out[i] = f2bf(in[i]);
}

// ---------- fragment load (works for global or LDS pointers; addrspace inferred) ----------
// 16-bit A/B 16x32 layout: lane l holds row (l&15); elems j<8: K=kb+8*(l>>4)+j,
// elems j>=8: K=kb+16+8*(l>>4)+(j-8).
__device__ __forceinline__ v16bf load_frag(const unsigned short* __restrict__ mat,
                                           int ldk, int rowTile, int kb, int lane) {
  int r  = rowTile + (lane & 15);
  int kh = (lane >> 4) << 3;
  const unsigned short* p = mat + (size_t)r * ldk + kb + kh;  // 16B aligned
  u16x8 lo = *(const u16x8*)p;
  u16x8 hi = *(const u16x8*)(p + 16);
  v16u u = __builtin_shufflevector(lo, hi, 0,1,2,3,4,5,6,7,8,9,10,11,12,13,14,15);
  return __builtin_bit_cast(v16bf, u);
}

// ---------- WMMA GEMM: C[M,N] = A[M,K] * W[N,K]^T (+bias, +relu) ----------
// Block = 128 thr (4 waves) -> 64 (M) x 64 (N) tile. Weight tile (NSUB*16 rows
// x K cols, bf16) staged into LDS once via the Tensor Data Mover; each wave
// computes a 16x(NSUB*16) strip, reusing one A fragment across NSUB WMMAs.
template <int K, int NSUB>
__global__ void wmma_gemm(const unsigned short* __restrict__ A,
                          const unsigned short* __restrict__ W,
                          const float* __restrict__ bias,
                          float* __restrict__ C,
                          unsigned short* __restrict__ Cbf,
                          int M, int Nn, int act) {
  __shared__ unsigned short wtile[64 * K];
  int lane = threadIdx.x & 31;
  int wave = threadIdx.x >> 5;
  int mrow = blockIdx.y * 64 + wave * 16;
  int n0   = blockIdx.x * 64;

  // ---- TDM stage: W[n0 .. n0+NSUB*16, 0..K) -> LDS (wave 0 issues) ----
  if (wave == 0) {
    unsigned long long ga = (unsigned long long)(const void*)(W + (size_t)n0 * K);
    unsigned lds = (unsigned)(size_t)(void*)wtile;
    u32x4 g0;
    g0[0] = 1u;                                           // count=1, user D#
    g0[1] = lds;                                          // lds_addr
    g0[2] = (unsigned)(ga & 0xffffffffu);                 // global_addr[31:0]
    g0[3] = (unsigned)((ga >> 32) & 0x01ffffffu) | (2u << 30);  // addr[56:32] | type=2
    i32x8 g1;
    g1[0] = (1 << 16);                                    // data_size=2B, no multicast
    g1[1] = (int)((K & 0xffff) << 16);                    // tensor_dim0[15:0]
    g1[2] = (int)(((unsigned)(Nn & 0xffff) << 16) | ((unsigned)K >> 16)); // td1 lo | td0 hi
    g1[3] = (int)(((unsigned)K << 16) | ((unsigned)Nn >> 16));            // tile_dim0=K | td1 hi
    g1[4] = NSUB * 16;                                    // tile_dim1=rows, tile_dim2=0
    g1[5] = K;                                            // tensor_dim0_stride[31:0]
    g1[6] = 0;                                            // stride hi | dim1_stride lo
    g1[7] = 0;
    i32x4 gz = {0, 0, 0, 0};
#if defined(__clang_major__) && __clang_major__ >= 23
    i32x8 gz8 = {0, 0, 0, 0, 0, 0, 0, 0};
    __builtin_amdgcn_tensor_load_to_lds(g0, g1, gz, gz, gz8, 0);
#else
    __builtin_amdgcn_tensor_load_to_lds(g0, g1, gz, gz, 0);
#endif
    __builtin_amdgcn_s_wait_tensorcnt(0);
  }
  __syncthreads();

  v8f zero = {0.f, 0.f, 0.f, 0.f, 0.f, 0.f, 0.f, 0.f};
  v8f acc[NSUB];
#pragma unroll
  for (int nn = 0; nn < NSUB; ++nn) acc[nn] = zero;

#pragma unroll 4
  for (int kb = 0; kb < K; kb += 32) {
    v16bf af = load_frag(A, K, mrow, kb, lane);          // global -> b128 loads
#pragma unroll
    for (int nn = 0; nn < NSUB; ++nn) {
      v16bf bf = load_frag(wtile, K, nn << 4, kb, lane); // LDS -> ds_load_b128
      acc[nn] = __builtin_amdgcn_wmma_f32_16x16x32_bf16(false, af, false, bf,
                                                        (short)0, acc[nn], false, false);
    }
  }

  // C/D layout: lane l -> col (l&15); VGPR r -> row mrow + r + 8*(l>>4)
  int mb = mrow + ((lane >> 4) << 3);
#pragma unroll
  for (int nn = 0; nn < NSUB; ++nn) {
    int n = n0 + (nn << 4) + (lane & 15);
    float bv = bias ? bias[n] : 0.0f;
#pragma unroll
    for (int r = 0; r < 8; ++r) {
      float v = acc[nn][r] + bv;
      if (act == 1) v = v > 0.f ? v : 0.f;
      size_t o = (size_t)(mb + r) * Nn + n;
      C[o] = v;
      if (Cbf) Cbf[o] = f2bf(v);
    }
  }
}

// ---------- LayerNorm (one wave / 256-elem row), optional residual + bf16 copy ----------
__global__ void ln_kernel(const float* __restrict__ x, const float* __restrict__ resid,
                          const float* __restrict__ g, const float* __restrict__ b,
                          float* __restrict__ out, unsigned short* __restrict__ outbf) {
  int lane = threadIdx.x & 31;
  int wave = threadIdx.x >> 5;
  int row  = blockIdx.x * 8 + wave;
  size_t base = (size_t)row * DIM + lane * 8;
  float v[8], s = 0.f, sq = 0.f;
#pragma unroll
  for (int j = 0; j < 8; ++j) {
    float t = x[base + j];
    if (resid) t += resid[base + j];
    v[j] = t; s += t; sq += t * t;
  }
#pragma unroll
  for (int m = 16; m >= 1; m >>= 1) { s += __shfl_xor(s, m, 32); sq += __shfl_xor(sq, m, 32); }
  float mean = s * (1.0f / DIM);
  float var  = sq * (1.0f / DIM) - mean * mean;
  float rst  = rsqrtf(var + 1e-5f);
#pragma unroll
  for (int j = 0; j < 8; ++j) {
    int c = lane * 8 + j;
    float o = (v[j] - mean) * rst * g[c] + b[c];
    out[base + j] = o;
    if (outbf) outbf[base + j] = f2bf(o);
  }
}

// ---------- GAT ----------
__global__ void gat_scores(const float* __restrict__ proj, const float* __restrict__ asrc,
                           const float* __restrict__ atgt, float* __restrict__ ssrc,
                           float* __restrict__ stgt) {
  int idx = blockIdx.x * blockDim.x + threadIdx.x;     // N*NH
  int n = idx >> 2, h = idx & 3;
  const float* p = proj + (size_t)n * DIM + h * FOUT_;
  float s1 = 0.f, s2 = 0.f;
#pragma unroll 8
  for (int f = 0; f < FOUT_; ++f) { float v = p[f]; s1 = fmaf(v, asrc[h*FOUT_+f], s1); s2 = fmaf(v, atgt[h*FOUT_+f], s2); }
  ssrc[idx] = s1; stgt[idx] = s2;
}

__global__ void init_gmax(int* g) { *g = (int)0x80000000; }

__global__ void edge_e(const int* __restrict__ src, const int* __restrict__ tgt,
                       const float* __restrict__ ssrc, const float* __restrict__ stgt,
                       float* __restrict__ ebuf, int* __restrict__ gmax, int E4) {
  int idx = blockIdx.x * blockDim.x + threadIdx.x;
  if (idx >= E4) return;
  int e = idx >> 2, h = idx & 3;
  float v = ssrc[src[e] * 4 + h] + stgt[tgt[e] * 4 + h];
  v = v > 0.f ? v : 0.2f * v;                          // leaky_relu(0.2)
  ebuf[idx] = v;
  int o = __float_as_int(v); o = o >= 0 ? o : (o ^ 0x7fffffff);
  atomicMax(gmax, o);
}

__global__ void edge_exp(const int* __restrict__ tgt, float* __restrict__ ebuf,
                         const int* __restrict__ gmax, float* __restrict__ denom, int E4) {
  int idx = blockIdx.x * blockDim.x + threadIdx.x;
  if (idx >= E4) return;
  int o = *gmax;
  float mx = __int_as_float(o >= 0 ? o : (o ^ 0x7fffffff));
  int e = idx >> 2, h = idx & 3;
  float ex = __expf(ebuf[idx] - mx);
  ebuf[idx] = ex;
  atomicAdd(&denom[tgt[e] * 4 + h], ex);
}

__global__ void edge_agg(const int* __restrict__ src, const int* __restrict__ tgt,
                         const float* __restrict__ ebuf, const float* __restrict__ denom,
                         const float* __restrict__ proj, float* __restrict__ agg, int E256) {
  int idx = blockIdx.x * blockDim.x + threadIdx.x;
  if (idx >= E256) return;
  int e = idx >> 8, r = idx & 255, h = r >> 6;
  float attn = ebuf[e * 4 + h] / (denom[tgt[e] * 4 + h] + 1e-16f);
  atomicAdd(&agg[(size_t)tgt[e] * DIM + r], proj[(size_t)src[e] * DIM + r] * attn);
}

__global__ void elu_res(float* __restrict__ agg, const float* __restrict__ x,
                        const float* __restrict__ gatb, int n) {
  int idx = blockIdx.x * blockDim.x + threadIdx.x;
  if (idx >= n) return;
  float v = agg[idx] + x[idx] + gatb[idx & 255];
  agg[idx] = v > 0.f ? v : (__expf(v) - 1.f);
}

// ---------- degree + stable argsort (u64 bitonic: key=(fbits<<32)|idx) ----------
__global__ void deg_kernel(const int* __restrict__ src, float* __restrict__ deg, int E) {
  int e = blockIdx.x * blockDim.x + threadIdx.x;
  if (e < E) atomicAdd(&deg[src[e]], 1.0f);
}
__global__ void key_kernel(const float* __restrict__ deg, const float* __restrict__ noise,
                           unsigned long long* __restrict__ key) {
  int i = blockIdx.x * blockDim.x + threadIdx.x;    // keys >= 0 -> bits order-preserving
  key[i] = ((unsigned long long)__float_as_uint(deg[i] + noise[i]) << 32) | (unsigned)i;
}
__global__ void bitonic_pass(unsigned long long* __restrict__ a, int j, int k) {
  int i = blockIdx.x * blockDim.x + threadIdx.x;
  int ixj = i ^ j;
  if (ixj > i) {
    unsigned long long ai = a[i], aj = a[ixj];
    if ((ai > aj) == ((i & k) == 0)) { a[i] = aj; a[ixj] = ai; }
  }
}
__global__ void perm_kernel(const unsigned long long* __restrict__ key,
                            int* __restrict__ perm, int* __restrict__ inv) {
  int i = blockIdx.x * blockDim.x + threadIdx.x;
  int p = (int)(key[i] & 0xffffffffu);
  perm[i] = p; inv[p] = i;
}
__global__ void gather_bf(const unsigned short* __restrict__ xbf, const int* __restrict__ perm,
                          unsigned short* __restrict__ out, int n) {
  int idx = blockIdx.x * blockDim.x + threadIdx.x;
  if (idx < n) out[idx] = xbf[(size_t)perm[idx >> 8] * DIM + (idx & 255)];
}

// ---------- Mamba ----------
__global__ void conv_silu(const float* __restrict__ xz, const float* __restrict__ cw,
                          const float* __restrict__ cb, float* __restrict__ xc,
                          unsigned short* __restrict__ xcbf, int n) {
  int idx = blockIdx.x * blockDim.x + threadIdx.x;
  if (idx >= n) return;
  int t = idx >> 8, d = idx & 255;
  float s = cb[d];
#pragma unroll
  for (int k = 0; k < 4; ++k) {
    int tt = t - 3 + k;
    float xv = tt >= 0 ? xz[(size_t)tt * 512 + d] : 0.f;
    s = fmaf(xv, cw[d * 4 + k], s);
  }
  float v = s * sigm(s);
  xc[idx] = v; xcbf[idx] = f2bf(v);
}

__global__ void dt_kernel(const float* __restrict__ dbc, const float* __restrict__ dtw,
                          const float* __restrict__ dtb, float* __restrict__ dt, int n) {
  int idx = blockIdx.x * blockDim.x + threadIdx.x;
  if (idx >= n) return;
  int t = idx >> 8, d = idx & 255;
  float acc = dtb[d];
#pragma unroll
  for (int r = 0; r < DTRANK; ++r) acc = fmaf(dbc[(size_t)t * 48 + r], dtw[d * 16 + r], acc);
  dt[idx] = acc > 20.f ? acc : log1pf(__expf(acc));    // softplus
}

// 4096 independent recurrences: lane = (d,s); shfl_xor sums over 16 states.
__global__ void mamba_scan(const float* __restrict__ dt, const float* __restrict__ xc,
                           const float* __restrict__ dbc, const float* __restrict__ alog,
                           float* __restrict__ y) {
  int gid = blockIdx.x * blockDim.x + threadIdx.x;   // 0..4095
  int d = gid >> 4, s = gid & 15;
  float A = -__expf(alog[d * DSTATE + s]);
  float h = 0.f;
  for (int t = 0; t < NODES; ++t) {
    if ((t & 63) == 0 && t + 64 < NODES) __builtin_prefetch(&dbc[(size_t)(t + 64) * 48], 0, 0);
    float dtv = dt[(size_t)t * DIM + d];
    float xv  = xc[(size_t)t * DIM + d];
    float Bv  = dbc[(size_t)t * 48 + DTRANK + s];
    float Cv  = dbc[(size_t)t * 48 + DTRANK + DSTATE + s];
    h = fmaf(__expf(dtv * A), h, dtv * xv * Bv);
    float c = h * Cv;
    c += __shfl_xor(c, 1, 32); c += __shfl_xor(c, 2, 32);
    c += __shfl_xor(c, 4, 32); c += __shfl_xor(c, 8, 32);
    if (s == 0) y[(size_t)t * DIM + d] = c;
  }
}

__global__ void ypost_kernel(const float* __restrict__ y, const float* __restrict__ xc,
                             const float* __restrict__ xz, const float* __restrict__ Dp,
                             unsigned short* __restrict__ out, int n) {
  int idx = blockIdx.x * blockDim.x + threadIdx.x;
  if (idx >= n) return;
  int t = idx >> 8, d = idx & 255;
  float v = y[idx] + xc[idx] * Dp[d];
  float z = xz[(size_t)t * 512 + 256 + d];
  out[idx] = f2bf(v * (z * sigm(z)));
}

__global__ void hattn_pre(const float* __restrict__ x, const float* __restrict__ ym,
                          const int* __restrict__ inv, float* __restrict__ pre, int n) {
  int idx = blockIdx.x * blockDim.x + threadIdx.x;
  if (idx < n) pre[idx] = x[idx] + ym[(size_t)inv[idx >> 8] * DIM + (idx & 255)];
}

__global__ void gate_combine(const float* __restrict__ gl, const float* __restrict__ gg,
                             const float* __restrict__ hl, const float* __restrict__ ha,
                             float* __restrict__ h, unsigned short* __restrict__ hbf, int n) {
  int idx = blockIdx.x * blockDim.x + threadIdx.x;
  if (idx >= n) return;
  float g = sigm(gl[idx] + gg[idx]);
  float v = g * hl[idx] + (1.f - g) * ha[idx];
  h[idx] = v; hbf[idx] = f2bf(v);
}

// ---------- host ----------
extern "C" void kernel_launch(void* const* d_in, const int* in_sizes, int n_in,
                              void* d_out, int out_size, void* d_ws, size_t ws_size,
                              hipStream_t stream) {
  const float* x      = (const float*)d_in[0];
  const int*   src    = (const int*)d_in[1];
  const int    E      = in_sizes[1] / 2;
  const int*   tgt    = src + E;
  const float* noise  = (const float*)d_in[3];
  const float* gatw   = (const float*)d_in[4];
  const float* asrc   = (const float*)d_in[5];
  const float* atgt   = (const float*)d_in[6];
  const float* gatb   = (const float*)d_in[7];
  const float* lnl_g  = (const float*)d_in[8];
  const float* lnl_b  = (const float*)d_in[9];
  const float* lng_g  = (const float*)d_in[10];
  const float* lng_b  = (const float*)d_in[11];
  const float* ln2_g  = (const float*)d_in[12];
  const float* ln2_b  = (const float*)d_in[13];
  const float* inw    = (const float*)d_in[14];
  const float* convw  = (const float*)d_in[15];
  const float* convb  = (const float*)d_in[16];
  const float* xpw    = (const float*)d_in[17];
  const float* dtw    = (const float*)d_in[18];
  const float* dtb    = (const float*)d_in[19];
  const float* alog   = (const float*)d_in[20];
  const float* Dp     = (const float*)d_in[21];
  const float* outw   = (const float*)d_in[22];
  const float* wl     = (const float*)d_in[23];
  const float* wg     = (const float*)d_in[24];
  const float* ff1w   = (const float*)d_in[25];
  const float* ff1b   = (const float*)d_in[26];
  const float* ff2w   = (const float*)d_in[27];
  const float* ff2b   = (const float*)d_in[28];
  float* out = (float*)d_out;

  char* ws = (char*)d_ws; size_t off = 0;
  auto alloc = [&](size_t bytes) -> void* {
    void* p = ws + off; off += (bytes + 255) & ~(size_t)255; return p;
  };
  const int ND = NODES * DIM;
  // bf16 buffers
  unsigned short* xbf    = (unsigned short*)alloc(ND * 2);
  unsigned short* gatWb  = (unsigned short*)alloc(256 * 256 * 2);
  unsigned short* inWb   = (unsigned short*)alloc(512 * 256 * 2);
  unsigned short* xpWb   = (unsigned short*)alloc(48 * 256 * 2);
  unsigned short* outWb  = (unsigned short*)alloc(256 * 256 * 2);
  unsigned short* wlWb   = (unsigned short*)alloc(256 * 256 * 2);
  unsigned short* wgWb   = (unsigned short*)alloc(256 * 256 * 2);
  unsigned short* ff1Wb  = (unsigned short*)alloc(512 * 256 * 2);
  unsigned short* ff2Wb  = (unsigned short*)alloc(256 * 512 * 2);
  unsigned short* xpermb = (unsigned short*)alloc(ND * 2);
  unsigned short* xcbf   = (unsigned short*)alloc(ND * 2);
  unsigned short* ypbf   = (unsigned short*)alloc(ND * 2);
  unsigned short* hlbf   = (unsigned short*)alloc(ND * 2);
  unsigned short* habf   = (unsigned short*)alloc(ND * 2);
  unsigned short* hbf    = (unsigned short*)alloc(ND * 2);
  unsigned short* ffhbf  = (unsigned short*)alloc(NODES * 512 * 2);
  // fp32 buffers
  float* proj  = (float*)alloc((size_t)ND * 4);
  float* ssrc  = (float*)alloc(NODES * 4 * 4);
  float* stgt  = (float*)alloc(NODES * 4 * 4);
  float* ebuf  = (float*)alloc((size_t)E * 4 * 4);
  float* denom = (float*)alloc(NODES * 4 * 4);
  float* agg   = (float*)alloc((size_t)ND * 4);
  float* hloc  = (float*)alloc((size_t)ND * 4);
  float* deg   = (float*)alloc(NODES * 4);
  float* xz    = (float*)alloc((size_t)NODES * 512 * 4);
  float* xc    = (float*)alloc((size_t)ND * 4);
  float* dbc   = (float*)alloc((size_t)NODES * 48 * 4);
  float* dt    = (float*)alloc((size_t)ND * 4);
  float* ysc   = (float*)alloc((size_t)ND * 4);
  float* ym    = (float*)alloc((size_t)ND * 4);
  float* hatt  = (float*)alloc((size_t)ND * 4);
  float* hcomb = (float*)alloc((size_t)ND * 4);
  float* ffh   = (float*)alloc((size_t)NODES * 512 * 4);
  float* ffo   = (float*)alloc((size_t)ND * 4);
  unsigned long long* keys = (unsigned long long*)alloc(NODES * 8);
  int* perm = (int*)alloc(NODES * 4);
  int* inv  = (int*)alloc(NODES * 4);
  int* gmax = (int*)alloc(256);
  float* hpre = proj;                 // reuse: proj dead after edge_agg
  float* gl = xz;                     // reuse: xz dead after ypost
  float* gg = xz + ND;

  const int TB = 256;
  auto ceil_ = [](int a, int b) { return (a + b - 1) / b; };

  // --- precision conversion ---
  cvt_bf16<<<ceil_(ND, TB), TB, 0, stream>>>(x, xbf, ND);
  cvt_bf16<<<ceil_(256*256, TB), TB, 0, stream>>>(gatw, gatWb, 256*256);
  cvt_bf16<<<ceil_(512*256, TB), TB, 0, stream>>>(inw, inWb, 512*256);
  cvt_bf16<<<ceil_(48*256, TB), TB, 0, stream>>>(xpw, xpWb, 48*256);
  cvt_bf16<<<ceil_(256*256, TB), TB, 0, stream>>>(outw, outWb, 256*256);
  cvt_bf16<<<ceil_(256*256, TB), TB, 0, stream>>>(wl, wlWb, 256*256);
  cvt_bf16<<<ceil_(256*256, TB), TB, 0, stream>>>(wg, wgWb, 256*256);
  cvt_bf16<<<ceil_(512*256, TB), TB, 0, stream>>>(ff1w, ff1Wb, 512*256);
  cvt_bf16<<<ceil_(256*512, TB), TB, 0, stream>>>(ff2w, ff2Wb, 256*512);

  // --- GAT branch ---
  wmma_gemm<256,4><<<dim3(4, NODES/64), 128, 0, stream>>>(xbf, gatWb, nullptr, proj, nullptr, NODES, 256, 0);
  gat_scores<<<ceil_(NODES*4, TB), TB, 0, stream>>>(proj, asrc, atgt, ssrc, stgt);
  init_gmax<<<1, 1, 0, stream>>>(gmax);
  hipMemsetAsync(denom, 0, NODES * 4 * 4, stream);
  hipMemsetAsync(agg, 0, (size_t)ND * 4, stream);
  edge_e<<<ceil_(E*4, TB), TB, 0, stream>>>(src, tgt, ssrc, stgt, ebuf, gmax, E*4);
  edge_exp<<<ceil_(E*4, TB), TB, 0, stream>>>(tgt, ebuf, gmax, denom, E*4);
  edge_agg<<<ceil_(E*256, TB), TB, 0, stream>>>(src, tgt, ebuf, denom, proj, agg, E*256);
  elu_res<<<ceil_(ND, TB), TB, 0, stream>>>(agg, x, gatb, ND);
  ln_kernel<<<NODES/8, 256, 0, stream>>>(agg, nullptr, lnl_g, lnl_b, hloc, hlbf);

  // --- degree sort ---
  hipMemsetAsync(deg, 0, NODES * 4, stream);
  deg_kernel<<<ceil_(E, TB), TB, 0, stream>>>(src, deg, E);
  key_kernel<<<NODES/TB, TB, 0, stream>>>(deg, noise, keys);
  for (int k = 2; k <= NODES; k <<= 1)
    for (int j = k >> 1; j > 0; j >>= 1)
      bitonic_pass<<<NODES/TB, TB, 0, stream>>>(keys, j, k);
  perm_kernel<<<NODES/TB, TB, 0, stream>>>(keys, perm, inv);
  gather_bf<<<ceil_(ND, TB), TB, 0, stream>>>(xbf, perm, xpermb, ND);

  // --- Mamba ---
  wmma_gemm<256,4><<<dim3(8, NODES/64), 128, 0, stream>>>(xpermb, inWb, nullptr, xz, nullptr, NODES, 512, 0);
  conv_silu<<<ceil_(ND, TB), TB, 0, stream>>>(xz, convw, convb, xc, xcbf, ND);
  wmma_gemm<256,3><<<dim3(1, NODES/64), 128, 0, stream>>>(xcbf, xpWb, nullptr, dbc, nullptr, NODES, 48, 0);
  dt_kernel<<<ceil_(ND, TB), TB, 0, stream>>>(dbc, dtw, dtb, dt, ND);
  mamba_scan<<<16, 256, 0, stream>>>(dt, xc, dbc, alog, ysc);
  ypost_kernel<<<ceil_(ND, TB), TB, 0, stream>>>(ysc, xc, xz, Dp, ypbf, ND);
  wmma_gemm<256,4><<<dim3(4, NODES/64), 128, 0, stream>>>(ypbf, outWb, nullptr, ym, nullptr, NODES, 256, 0);
  hattn_pre<<<ceil_(ND, TB), TB, 0, stream>>>(x, ym, inv, hpre, ND);
  ln_kernel<<<NODES/8, 256, 0, stream>>>(hpre, nullptr, lng_g, lng_b, hatt, habf);

  // --- gate + FFN ---
  wmma_gemm<256,4><<<dim3(4, NODES/64), 128, 0, stream>>>(hlbf, wlWb, nullptr, gl, nullptr, NODES, 256, 0);
  wmma_gemm<256,4><<<dim3(4, NODES/64), 128, 0, stream>>>(habf, wgWb, nullptr, gg, nullptr, NODES, 256, 0);
  gate_combine<<<ceil_(ND, TB), TB, 0, stream>>>(gl, gg, hloc, hatt, hcomb, hbf, ND);
  wmma_gemm<256,4><<<dim3(8, NODES/64), 128, 0, stream>>>(hbf, ff1Wb, ff1b, ffh, ffhbf, NODES, 512, 1);
  wmma_gemm<512,4><<<dim3(4, NODES/64), 128, 0, stream>>>(ffhbf, ff2Wb, ff2b, ffo, nullptr, NODES, 256, 0);
  ln_kernel<<<NODES/8, 256, 0, stream>>>(hcomb, ffo, ln2_g, ln2_b, out, nullptr);

  (void)n_in; (void)out_size; (void)ws_size;
}
